// computeHausdorffLoss_17446157156870
// MI455X (gfx1250) — matrix operand order repaired and checked by reading, not compiled, not run on hardware
//
#include <hip/hip_runtime.h>
#include <hip/hip_bf16.h>
#include <math.h>

// ---------------------------------------------------------------------------
// Hausdorff-style max-pair-distance loss, MI455X (gfx1250, wave32, WMMA).
//
// Exactness: d2(p,q) is convex in each coordinate -> per column only min-row /
// max-row masked pixels can achieve the max -> <=256 candidates per mask per
// image. d2 evaluated as an exact f16 dot product (all encoded values are
// integers <= 256; f16 products exact; f32 accumulation exact) with
// v_wmma_f32_16x16x32_f16 on 16x16 candidate tiles.
// ---------------------------------------------------------------------------

typedef __attribute__((ext_vector_type(16))) _Float16 v16h;
typedef __attribute__((ext_vector_type(8)))  _Float16 v8h;
typedef __attribute__((ext_vector_type(8)))  float    v8f;

#define BATCH  4
#define HDIM   128
#define WDIM   128
#define PIX    (HDIM * WDIM)
#define NCAND  256            // 2 candidates (min/max row) per column
#define KSL    16             // K slots stored per candidate in global ws
#define KPAD   32             // K slots per candidate row in LDS (upper 16 = 0)

// K-slot encoding (a = mask1 candidate p, b = mask2 candidate q):
//  0: ci_p       , -2*ci_q      ->  -2 ci_p ci_q
//  1: cj_p       , -2*cj_q      ->  -2 cj_p cj_q
//  2: ci_p^2>>8  , 256          \   ci_p^2
//  3: ci_p^2&255 , 1            /
//  4: cj_p^2>>8  , 256          \   cj_p^2
//  5: cj_p^2&255 , 1            /
//  6: 256        , ci_q^2>>8    \   ci_q^2
//  7: 1          , ci_q^2&255   /
//  8: 256        , cj_q^2>>8    \   cj_q^2
//  9: 1          , cj_q^2&255   /
// 10: invA?256:0 , -255         ->  invalid-A sentinel (-65280)
// 11: -255       , invB?256:0   ->  invalid-B sentinel (-65280)
// valid x valid = exact d2 in [0, 32258]; any invalid side <= -33022.

__global__ void hd_candidates_kernel(const float* __restrict__ outp,
                                     const float* __restrict__ targ,
                                     _Float16* __restrict__ aVec,
                                     _Float16* __restrict__ bVec) {
  __shared__ int s_min[WDIM];
  __shared__ int s_max[WDIM];
  const int b   = blockIdx.x >> 1;
  const int s   = blockIdx.x & 1;          // 0: mask1 (output>0.5), 1: mask2
  const int tid = threadIdx.x;             // 256 threads

  const float* src = (s == 0 ? outp : targ) + (size_t)b * PIX;
  __builtin_prefetch(src + tid * 64, 0, 1);  // global_prefetch_b8, warm L2

  if (tid < WDIM) { s_min[tid] = 0x7FFFFFFF; s_max[tid] = -1; }
  __syncthreads();

  for (int e = tid; e < PIX; e += 256) {
    float v = src[e];
    bool m = (s == 0) ? (v > 0.5f) : (v != 0.0f);
    if (m) {
      int i = e >> 7, j = e & (WDIM - 1);
      atomicMin(&s_min[j], i);             // LDS atomics (ds_min/ds_max)
      atomicMax(&s_max[j], i);
    }
  }
  __syncthreads();

  // one candidate vector per thread
  const int j      = tid >> 1;
  const bool useMx = tid & 1;
  const bool valid = (s_max[j] >= 0);
  int row = useMx ? s_max[j] : s_min[j];
  if (!valid) row = 0;
  const int sqi = row * row, sqj = j * j;

  _Float16 vec[KSL];
#pragma unroll
  for (int k = 0; k < KSL; ++k) vec[k] = (_Float16)0.0f;

  if (s == 0) {                             // A side
    if (valid) {
      vec[0] = (_Float16)(float)row;
      vec[1] = (_Float16)(float)j;
      vec[2] = (_Float16)(float)(sqi >> 8);
      vec[3] = (_Float16)(float)(sqi & 255);
      vec[4] = (_Float16)(float)(sqj >> 8);
      vec[5] = (_Float16)(float)(sqj & 255);
      vec[6] = (_Float16)256.0f; vec[7] = (_Float16)1.0f;
      vec[8] = (_Float16)256.0f; vec[9] = (_Float16)1.0f;
    } else {
      vec[10] = (_Float16)256.0f;
    }
    vec[11] = (_Float16)(-255.0f);
    _Float16* dst = aVec + ((size_t)b * NCAND + tid) * KSL;
#pragma unroll
    for (int k = 0; k < KSL; ++k) dst[k] = vec[k];
  } else {                                  // B side
    if (valid) {
      vec[0] = (_Float16)(-2.0f * (float)row);
      vec[1] = (_Float16)(-2.0f * (float)j);
      vec[2] = (_Float16)256.0f; vec[3] = (_Float16)1.0f;
      vec[4] = (_Float16)256.0f; vec[5] = (_Float16)1.0f;
      vec[6] = (_Float16)(float)(sqi >> 8);
      vec[7] = (_Float16)(float)(sqi & 255);
      vec[8] = (_Float16)(float)(sqj >> 8);
      vec[9] = (_Float16)(float)(sqj & 255);
    } else {
      vec[11] = (_Float16)256.0f;
    }
    vec[10] = (_Float16)(-255.0f);
    _Float16* dst = bVec + ((size_t)b * NCAND + tid) * KSL;
#pragma unroll
    for (int k = 0; k < KSL; ++k) dst[k] = vec[k];
  }
}

// One block (8 waves) per image: 16x16 tile-pairs of 16x16 WMMA over the
// 256x256 candidate distance matrix. LDS candidate rows are padded to K=32
// with zeros so every lane builds its fragment from two contiguous
// ds_load_b128's -- no divergent control flow, no zero-mov chains.
__global__ void hd_wmma_max_kernel(const _Float16* __restrict__ aVec,
                                   const _Float16* __restrict__ bVec,
                                   float* __restrict__ bmax) {
  __shared__ _Float16 sA[NCAND * KPAD];    // 16 KB, rows: K0..15 data, K16..31 = 0
  __shared__ _Float16 sB[NCAND * KPAD];    // 16 KB
  __shared__ float    s_wmax[8];

  const int b   = blockIdx.x;
  const int tid = threadIdx.x;             // 256 threads = 8 waves

  // stage + zero-pad: each LDS row = 4 x uint4 {data0, data1, 0, 0}
  const uint4* gA = (const uint4*)(aVec + (size_t)b * NCAND * KSL);
  const uint4* gB = (const uint4*)(bVec + (size_t)b * NCAND * KSL);
  uint4* lA = (uint4*)sA;
  uint4* lB = (uint4*)sB;
  const uint4 z4 = {0u, 0u, 0u, 0u};
  for (int e = tid; e < NCAND * 2; e += 256) {   // 512 data chunks
    int r = e >> 1, q = e & 1;
    lA[r * 4 + q] = gA[e];
    lB[r * 4 + q] = gB[e];
  }
  for (int e = tid; e < NCAND * 2; e += 256) {   // 512 zero chunks
    int r = e >> 1, q = e & 1;
    lA[r * 4 + 2 + q] = z4;
    lB[r * 4 + 2 + q] = z4;
  }
  __syncthreads();

  const int wave  = tid >> 5;
  const int lane  = tid & 31;
  const int lhalf = lane & 15;
  const int hi    = (lane >> 4) & 1;       // 0: lanes 0-15, 1: lanes 16-31

  v8f vmax;
#pragma unroll
  for (int k = 0; k < 8; ++k) vmax[k] = -3.0e38f;

  for (int mi = 0; mi < 2; ++mi) {
    const int mt = wave * 2 + mi;          // this wave's M tile (A candidates)
    // A fragment 16x32: lane row M = lhalf; chunks of 8 K-values per v8h:
    // lo lanes take K{0..7} and K{16..23}; hi lanes K{8..15} and K{24..31}.
    const v8h* ar = (const v8h*)(sA + (mt * 16 + lhalf) * KPAD);
    v8h alo = ar[hi];
    v8h ahi = ar[2 + hi];
    v16h afrag = __builtin_shufflevector(alo, ahi,
        0, 1, 2, 3, 4, 5, 6, 7, 8, 9, 10, 11, 12, 13, 14, 15);

#pragma unroll 4
    for (int nt = 0; nt < 16; ++nt) {
      // B fragment 32x16: lane column N = lhalf; same K order as A:
      // lo lanes K{0..7} and K{16..23}; hi lanes K{8..15} and K{24..31}.
      const v8h* br = (const v8h*)(sB + (nt * 16 + lhalf) * KPAD);
      v8h blo = br[hi];
      v8h bhi = br[2 + hi];
      v16h bfrag = __builtin_shufflevector(blo, bhi,
          0, 1, 2, 3, 4, 5, 6, 7, 8, 9, 10, 11, 12, 13, 14, 15);

      v8f c = {};
      v8f d = __builtin_amdgcn_wmma_f32_16x16x32_f16(
          /*neg_a=*/false, afrag, /*neg_b=*/false, bfrag,
          /*c_mod=*/(short)0, c, /*reuse_a=*/false, /*reuse_b=*/false);
#pragma unroll
      for (int k = 0; k < 8; ++k) vmax[k] = fmaxf(vmax[k], d[k]);
    }
  }

  // horizontal 8 -> 1, then wave32 max reduction
  float runmax = vmax[0];
#pragma unroll
  for (int k = 1; k < 8; ++k) runmax = fmaxf(runmax, vmax[k]);
#pragma unroll
  for (int off = 16; off >= 1; off >>= 1)
    runmax = fmaxf(runmax, __shfl_xor(runmax, off, 32));
  if (lane == 0) s_wmax[wave] = runmax;
  __syncthreads();
  if (tid == 0) {
    float m = s_wmax[0];
#pragma unroll
    for (int w = 1; w < 8; ++w) m = fmaxf(m, s_wmax[w]);
    bmax[b] = m;
  }
}

__global__ void hd_finalize_kernel(const float* __restrict__ bmax,
                                   float* __restrict__ out) {
  if (threadIdx.x == 0 && blockIdx.x == 0) {
    float acc = 0.0f;
#pragma unroll
    for (int b = 0; b < BATCH; ++b) {
      float md2 = bmax[b];
      // any negative max <=> no valid (p,q) pair <=> +inf (torch semantics)
      float hd = (md2 < -0.5f) ? __builtin_huge_valf()
                               : sqrtf(fmaxf(md2, 0.0f));
      acc += hd;
    }
    out[0] = acc / (float)BATCH;
  }
}

extern "C" void kernel_launch(void* const* d_in, const int* in_sizes, int n_in,
                              void* d_out, int out_size, void* d_ws, size_t ws_size,
                              hipStream_t stream) {
  const float* outp = (const float*)d_in[0];   // output: (4,128,128) f32
  const float* targ = (const float*)d_in[1];   // target: (4,128,128) f32

  _Float16* aVec = (_Float16*)d_ws;                      // 4*256*16 halfs = 32 KB
  _Float16* bVec = aVec + (size_t)BATCH * NCAND * KSL;   // 32 KB
  float*    bmax = (float*)(bVec + (size_t)BATCH * NCAND * KSL); // 16 B

  hd_candidates_kernel<<<BATCH * 2, 256, 0, stream>>>(outp, targ, aVec, bVec);
  hd_wmma_max_kernel<<<BATCH, 256, 0, stream>>>(aVec, bVec, bmax);
  hd_finalize_kernel<<<1, 64, 0, stream>>>(bmax, (float*)d_out);
}
